// FGCF_42992622633207
// MI455X (gfx1250) — compile-verified, matrix operand order checked by reading, not run on hardware
//
#include <hip/hip_runtime.h>
#include <hip/hip_bf16.h>
#include <math.h>

#define NU 512
#define NI 1024
#define ND 64

typedef __attribute__((ext_vector_type(16))) _Float16 v16h;
typedef __attribute__((ext_vector_type(8)))  float    v8f;

// ---------------------------------------------------------------------------
// 1. attention scores  w[u,i] = tanh(se . W1 + b1) . W2 + b2
//    Streams the 128 MB ui_table gather once (HBM-bound stage).
// ---------------------------------------------------------------------------
__global__ void att_kernel(const int* __restrict__ u_v_idx,
                           const float* __restrict__ ui_table,
                           const float* __restrict__ W1, const float* __restrict__ b1,
                           const float* __restrict__ W2, const float* __restrict__ b2,
                           float* __restrict__ w_att)
{
    __shared__ float sW1[ND * 8];
    __shared__ float sb1[8];
    __shared__ float sW2[8];
    int t = threadIdx.x;
    for (int j = t; j < ND * 8; j += blockDim.x) sW1[j] = W1[j];
    if (t < 8) { sb1[t] = b1[t]; sW2[t] = W2[t]; }
    __syncthreads();

    size_t idx = (size_t)blockIdx.x * blockDim.x + t;
    const float* se = ui_table + (size_t)u_v_idx[idx] * ND;
    __builtin_prefetch(se, 0, 1);   // global_prefetch_b8 (gfx1250)

    float acc[8];
#pragma unroll
    for (int k = 0; k < 8; ++k) acc[k] = 0.0f;
#pragma unroll
    for (int d4 = 0; d4 < ND / 4; ++d4) {
        float4 x = ((const float4*)se)[d4];
        int d = d4 * 4;
#pragma unroll
        for (int k = 0; k < 8; ++k) {
            acc[k] += x.x * sW1[(d + 0) * 8 + k];
            acc[k] += x.y * sW1[(d + 1) * 8 + k];
            acc[k] += x.z * sW1[(d + 2) * 8 + k];
            acc[k] += x.w * sW1[(d + 3) * 8 + k];
        }
    }
    float w = b2[0];
#pragma unroll
    for (int k = 0; k < 8; ++k) w += tanhf(acc[k] + sb1[k]) * sW2[k];
    w_att[idx] = w;
}

// ---------------------------------------------------------------------------
// 2. per-user softmax over items; only beta[:, I-1] is used:
//    s2d[u] = I * se[u, I-1, :] * beta_last + ue[u]
// ---------------------------------------------------------------------------
__global__ void softmax_s2d_kernel(const float* __restrict__ w_att,
                                   const int* __restrict__ u_v_idx,
                                   const float* __restrict__ ui_table,
                                   const int* __restrict__ user_idx,
                                   const float* __restrict__ user_table,
                                   float* __restrict__ s2d)
{
    __shared__ float red[256];
    int u = blockIdx.x;
    int t = threadIdx.x;
    const float* wrow = w_att + (size_t)u * NI;

    float m = -1e30f;
    for (int i = t; i < NI; i += 256) m = fmaxf(m, wrow[i]);
    red[t] = m; __syncthreads();
    for (int s = 128; s > 0; s >>= 1) { if (t < s) red[t] = fmaxf(red[t], red[t + s]); __syncthreads(); }
    float mx = red[0]; __syncthreads();

    float sum = 0.0f;
    for (int i = t; i < NI; i += 256) sum += expf(wrow[i] - mx);
    red[t] = sum; __syncthreads();
    for (int s = 128; s > 0; s >>= 1) { if (t < s) red[t] += red[t + s]; __syncthreads(); }
    float total = red[0];

    float beta_last = expf(wrow[NI - 1] - mx) / total;
    if (t < ND) {
        const float* se_last = ui_table + (size_t)u_v_idx[(size_t)u * NI + (NI - 1)] * ND;
        float ue = user_table[(size_t)user_idx[u] * ND + t];
        s2d[(size_t)u * ND + t] = (float)NI * se_last[t] * beta_last + ue;
    }
}

// ---------------------------------------------------------------------------
// 3. GCN normalization degree helpers
// ---------------------------------------------------------------------------
__global__ void deg_init_kernel(float* __restrict__ deg, int n)
{
    int i = blockIdx.x * blockDim.x + threadIdx.x;
    if (i < n) deg[i] = 1.0f;   // self loop
}
__global__ void deg_add_kernel(const int* __restrict__ dst, float* __restrict__ deg, int ne)
{
    int i = blockIdx.x * blockDim.x + threadIdx.x;
    if (i < ne) atomicAdd(&deg[dst[i]], 1.0f);
}
__global__ void deg_fin_kernel(float* __restrict__ deg, int n)
{
    int i = blockIdx.x * blockDim.x + threadIdx.x;
    if (i < n) deg[i] = rsqrtf(deg[i]);
}

// ---------------------------------------------------------------------------
// 4. Branch-free f32 GEMM via v_wmma_f32_16x16x32_f16.
//    PRECONDITION: K % 32 == 0 (caller pre-pads), M % 16 == 0, N % 16 == 0.
//    One wave per 16x16 C tile.  A-side: 4x float4 (b128) loads per K-chunk.
//    16-bit A 16x32 VGPR layout: elem jj -> K = kc + (jj<8?0:16) + g*8 + jj%8
// ---------------------------------------------------------------------------
__global__ void gemm_wmma_f16(const float* __restrict__ A, int lda,
                              const float* __restrict__ B, int ldb,
                              float* __restrict__ C, int ldc, int K)
{
    int m0 = blockIdx.x * 16;
    int n0 = blockIdx.y * 16;
    int lane = threadIdx.x & 31;
    int g  = lane >> 4;
    int mr = lane & 15;

    const float* arow = A + (m0 + mr) * lda + g * 8;
    const float* bcol = B + (n0 + mr) + g * 8 * ldb;

    v8f acc = {};
    for (int kc = 0; kc < K; kc += 32) {
        float4 a0 = *(const float4*)(arow + kc);
        float4 a1 = *(const float4*)(arow + kc + 4);
        float4 a2 = *(const float4*)(arow + kc + 16);
        float4 a3 = *(const float4*)(arow + kc + 20);
        v16h a;
        a[0]  = (_Float16)a0.x; a[1]  = (_Float16)a0.y; a[2]  = (_Float16)a0.z; a[3]  = (_Float16)a0.w;
        a[4]  = (_Float16)a1.x; a[5]  = (_Float16)a1.y; a[6]  = (_Float16)a1.z; a[7]  = (_Float16)a1.w;
        a[8]  = (_Float16)a2.x; a[9]  = (_Float16)a2.y; a[10] = (_Float16)a2.z; a[11] = (_Float16)a2.w;
        a[12] = (_Float16)a3.x; a[13] = (_Float16)a3.y; a[14] = (_Float16)a3.z; a[15] = (_Float16)a3.w;

        v16h b;
#pragma unroll
        for (int jj = 0; jj < 16; ++jj) {
            int krel = kc + ((jj & 8) ? 16 : 0) + (jj & 7);
            b[jj] = (_Float16)bcol[krel * ldb];
        }
        acc = __builtin_amdgcn_wmma_f32_16x16x32_f16(false, a, false, b,
                                                     (short)0, acc, false, false);
    }
#pragma unroll
    for (int r = 0; r < 8; ++r) {
        int row = m0 + r + g * 8;
        int col = n0 + mr;
        C[(size_t)row * ldc + col] = acc[r];
    }
}

// ---------------------------------------------------------------------------
// 5. GCN aggregation: out[v] = sum_{(s->v)} h[s]*dinv[s]*dinv[v]  (+ self loop)
// ---------------------------------------------------------------------------
__global__ void agg_init_kernel(const float* __restrict__ h, const float* __restrict__ dinv,
                                float* __restrict__ out, int n, int F)
{
    int idx = blockIdx.x * blockDim.x + threadIdx.x;
    if (idx >= n * F) return;
    int node = idx / F;
    float di = dinv[node];
    out[idx] = h[idx] * di * di;
}
__global__ void agg_edge_kernel(const float* __restrict__ h,
                                const int* __restrict__ src, const int* __restrict__ dst,
                                const float* __restrict__ dinv,
                                float* __restrict__ out, int ne, int F)
{
    int idx = blockIdx.x * blockDim.x + threadIdx.x;
    if (idx >= ne * F) return;
    int e = idx / F, f = idx - e * F;
    int s = src[e], d = dst[e];
    atomicAdd(&out[d * F + f], h[s * F + f] * dinv[s] * dinv[d]);
}
__global__ void agg_fin_kernel(float* __restrict__ out, const float* __restrict__ b, int n, int F)
{
    int idx = blockIdx.x * blockDim.x + threadIdx.x;
    if (idx >= n * F) return;
    out[idx] = fmaxf(out[idx] + b[idx % F], 0.0f);
}

// ---------------------------------------------------------------------------
// 6. misc: gather pe, concat(+pad) eu/el, weight pad, f32 -> f16 convert
// ---------------------------------------------------------------------------
__global__ void gather_pe_kernel(const int* __restrict__ item_idx,
                                 const float* __restrict__ item_table,
                                 float* __restrict__ pe)
{
    int idx = blockIdx.x * blockDim.x + threadIdx.x;
    if (idx >= NI * ND) return;
    int i = idx / ND, d = idx - i * ND;
    pe[idx] = item_table[(size_t)item_idx[i] * ND + d];
}

// writes K-padded (stride 288, zeros at 272..287) copy AND the dense 272 output
__global__ void concat_pad_kernel(const float* __restrict__ a1 /*64*/,
                                  const float* __restrict__ a2 /*64*/,
                                  const float* __restrict__ a3 /*32*/,
                                  const float* __restrict__ a4 /*16*/,
                                  float* __restrict__ outP,    // n x 288
                                  float* __restrict__ out272,  // n x 272 (d_out slice)
                                  int n)
{
    // layout: [a1(64), a2(64), a3(32), a1(64), a4(16), a3(32)] = 272, pad to 288
    int idx = blockIdx.x * blockDim.x + threadIdx.x;
    if (idx >= n * 288) return;
    int r = idx / 288, f = idx - r * 288;
    float v = 0.0f;
    if      (f < 64)  v = a1[r * 64 + f];
    else if (f < 128) v = a2[r * 64 + (f - 64)];
    else if (f < 160) v = a3[r * 32 + (f - 128)];
    else if (f < 224) v = a1[r * 64 + (f - 160)];
    else if (f < 240) v = a4[r * 16 + (f - 224)];
    else if (f < 272) v = a3[r * 32 + (f - 240)];
    outP[idx] = v;
    if (f < 272) out272[r * 272 + f] = v;
}

// lin1_W (544x128) -> two zero-padded halves Wp1/Wp2 (288x128 each)
__global__ void pad_w_kernel(const float* __restrict__ lin1_W,
                             float* __restrict__ Wp1, float* __restrict__ Wp2)
{
    int idx = blockIdx.x * blockDim.x + threadIdx.x;
    if (idx >= 288 * 128) return;
    int k = idx / 128, c = idx - k * 128;
    float v1 = 0.0f, v2 = 0.0f;
    if (k < 272) { v1 = lin1_W[k * 128 + c]; v2 = lin1_W[(k + 272) * 128 + c]; }
    Wp1[idx] = v1;
    Wp2[idx] = v2;
}

__global__ void cvt_f16_kernel(const float* __restrict__ src, _Float16* __restrict__ dst, int n)
{
    int i = blockIdx.x * blockDim.x + threadIdx.x;
    if (i < n) dst[i] = (_Float16)src[i];
}

// ---------------------------------------------------------------------------
// 7. pair MLP:  h1 = relu(pre_u[u] + pre_v[i] + b1)         (128)
//               h2 = relu(h1 @ lin2_W + b2)                 (64)  <- WMMA x4
//               score = relu(h2 . out_W + out_b)
//    block = 128 threads = 4 waves; lin2_W staged in LDS (16 KB).
// ---------------------------------------------------------------------------
__global__ void pair_mlp_kernel(const float* __restrict__ pre_u,
                                const float* __restrict__ pre_v,
                                const float* __restrict__ lin1_b,
                                const _Float16* __restrict__ Bh,   // 128x64 f16
                                const float* __restrict__ lin2_b,
                                const float* __restrict__ out_W,
                                const float* __restrict__ out_b,
                                float* __restrict__ score)
{
    __shared__ _Float16 sB[128 * 64];
    for (int j = threadIdx.x; j < (128 * 64) / 8; j += 128)
        ((float4*)sB)[j] = ((const float4*)Bh)[j];
    __syncthreads();

    int wid  = threadIdx.x >> 5;
    int lane = threadIdx.x & 31;
    int g  = lane >> 4;
    int mr = lane & 15;
    int u  = blockIdx.x;
    int i0 = (blockIdx.y * 4 + wid) * 16;

    const float* pu = pre_u + u * 128 + g * 8;
    const float* pv = pre_v + (i0 + mr) * 128 + g * 8;
    const float* pb = lin1_b + g * 8;

    v8f acc0 = {}, acc1 = {}, acc2 = {}, acc3 = {};
#pragma unroll
    for (int kc = 0; kc < 128; kc += 32) {
        float4 u0 = *(const float4*)(pu + kc);      float4 u1 = *(const float4*)(pu + kc + 4);
        float4 u2 = *(const float4*)(pu + kc + 16); float4 u3 = *(const float4*)(pu + kc + 20);
        float4 v0 = *(const float4*)(pv + kc);      float4 v1 = *(const float4*)(pv + kc + 4);
        float4 v2 = *(const float4*)(pv + kc + 16); float4 v3 = *(const float4*)(pv + kc + 20);
        float4 b0 = *(const float4*)(pb + kc);      float4 b1 = *(const float4*)(pb + kc + 4);
        float4 b2 = *(const float4*)(pb + kc + 16); float4 b3 = *(const float4*)(pb + kc + 20);
        v16h a;
        a[0]  = (_Float16)fmaxf(u0.x + v0.x + b0.x, 0.0f);
        a[1]  = (_Float16)fmaxf(u0.y + v0.y + b0.y, 0.0f);
        a[2]  = (_Float16)fmaxf(u0.z + v0.z + b0.z, 0.0f);
        a[3]  = (_Float16)fmaxf(u0.w + v0.w + b0.w, 0.0f);
        a[4]  = (_Float16)fmaxf(u1.x + v1.x + b1.x, 0.0f);
        a[5]  = (_Float16)fmaxf(u1.y + v1.y + b1.y, 0.0f);
        a[6]  = (_Float16)fmaxf(u1.z + v1.z + b1.z, 0.0f);
        a[7]  = (_Float16)fmaxf(u1.w + v1.w + b1.w, 0.0f);
        a[8]  = (_Float16)fmaxf(u2.x + v2.x + b2.x, 0.0f);
        a[9]  = (_Float16)fmaxf(u2.y + v2.y + b2.y, 0.0f);
        a[10] = (_Float16)fmaxf(u2.z + v2.z + b2.z, 0.0f);
        a[11] = (_Float16)fmaxf(u2.w + v2.w + b2.w, 0.0f);
        a[12] = (_Float16)fmaxf(u3.x + v3.x + b3.x, 0.0f);
        a[13] = (_Float16)fmaxf(u3.y + v3.y + b3.y, 0.0f);
        a[14] = (_Float16)fmaxf(u3.z + v3.z + b3.z, 0.0f);
        a[15] = (_Float16)fmaxf(u3.w + v3.w + b3.w, 0.0f);

#define PAIR_NT(ACC, NT)                                                          \
        {                                                                         \
            v16h b;                                                               \
            _Pragma("unroll")                                                     \
            for (int jj = 0; jj < 16; ++jj) {                                     \
                int k = kc + ((jj & 8) ? 16 : 0) + g * 8 + (jj & 7);              \
                b[jj] = sB[k * 64 + (NT) * 16 + mr];                              \
            }                                                                     \
            ACC = __builtin_amdgcn_wmma_f32_16x16x32_f16(false, a, false, b,      \
                                                         (short)0, ACC, false, false); \
        }
        PAIR_NT(acc0, 0)
        PAIR_NT(acc1, 1)
        PAIR_NT(acc2, 2)
        PAIR_NT(acc3, 3)
#undef PAIR_NT
    }

    float part[8];
#pragma unroll
    for (int r = 0; r < 8; ++r) part[r] = 0.0f;
#define PAIR_EPI(ACC, NT)                                                         \
    {                                                                             \
        int col = (NT) * 16 + mr;                                                 \
        float lb = lin2_b[col];                                                   \
        float ow = out_W[col];                                                    \
        _Pragma("unroll")                                                         \
        for (int r = 0; r < 8; ++r)                                               \
            part[r] += fmaxf(ACC[r] + lb, 0.0f) * ow;                             \
    }
    PAIR_EPI(acc0, 0)
    PAIR_EPI(acc1, 1)
    PAIR_EPI(acc2, 2)
    PAIR_EPI(acc3, 3)
#undef PAIR_EPI

#pragma unroll
    for (int mask = 1; mask < 16; mask <<= 1) {
#pragma unroll
        for (int r = 0; r < 8; ++r)
            part[r] += __shfl_xor(part[r], mask, 32);
    }
    if (mr == 0) {
        float ob = out_b[0];
#pragma unroll
        for (int r = 0; r < 8; ++r) {
            int item = i0 + r + g * 8;
            score[(size_t)u * NI + item] = fmaxf(part[r] + ob, 0.0f);
        }
    }
}

// ---------------------------------------------------------------------------
// launch
// ---------------------------------------------------------------------------
extern "C" void kernel_launch(void* const* d_in, const int* in_sizes, int n_in,
                              void* d_out, int out_size, void* d_ws, size_t ws_size,
                              hipStream_t stream)
{
    (void)in_sizes; (void)n_in; (void)out_size; (void)ws_size;

    const int*   u_v_idx    = (const int*)d_in[0];
    const int*   user_idx   = (const int*)d_in[1];
    const int*   item_idx   = (const int*)d_in[2];
    const int*   ug_u_edges = (const int*)d_in[3];   // (2, NU*32)
    const int*   ug_v_edges = (const int*)d_in[4];   // (2, NI*32)
    const float* ui_table   = (const float*)d_in[5];
    const float* user_table = (const float*)d_in[6];
    const float* item_table = (const float*)d_in[7];
    const float* att_W1 = (const float*)d_in[8];
    const float* att_b1 = (const float*)d_in[9];
    const float* att_W2 = (const float*)d_in[10];
    const float* att_b2 = (const float*)d_in[11];
    const float* ugcn_W1 = (const float*)d_in[12]; const float* ugcn_b1 = (const float*)d_in[13];
    const float* ugcn_W2 = (const float*)d_in[14]; const float* ugcn_b2 = (const float*)d_in[15];
    const float* ugcn_W3 = (const float*)d_in[16]; const float* ugcn_b3 = (const float*)d_in[17];
    const float* igcn_W1 = (const float*)d_in[18]; const float* igcn_b1 = (const float*)d_in[19];
    const float* igcn_W2 = (const float*)d_in[20]; const float* igcn_b2 = (const float*)d_in[21];
    const float* igcn_W3 = (const float*)d_in[22]; const float* igcn_b3 = (const float*)d_in[23];
    const float* lin1_W = (const float*)d_in[24]; const float* lin1_b = (const float*)d_in[25];
    const float* lin2_W = (const float*)d_in[26]; const float* lin2_b = (const float*)d_in[27];
    const float* out_W  = (const float*)d_in[28]; const float* out_b  = (const float*)d_in[29];

    float* out = (float*)d_out;
    float* ws  = (float*)d_ws;

    // workspace layout (float offsets)
    float* w_att  = ws;               // 524288
    float* s2d    = ws + 524288;      // NU*64
    float* eu1    = ws + 557056;      // NU*64
    float* eu2    = ws + 589824;      // NU*32
    float* eu3    = ws + 606208;      // NU*16
    float* el1    = ws + 614400;      // NI*64
    float* el2    = ws + 679936;      // NI*32
    float* el3    = ws + 712704;      // NI*16
    float* dinv_u = ws + 729088;      // NU
    float* dinv_v = ws + 729600;      // NI
    float* htmp   = ws + 730624;      // NI*64
    float* pre_u  = ws + 796160;      // NU*128
    float* pre_v  = ws + 861696;      // NI*128 (holds pe early)
    _Float16* lin2h = (_Float16*)(ws + 992768); // 128*64 halves
    float* euP    = ws + 996864;      // NU*288 (K-padded eu)
    float* elP    = ws + 1144320;     // NI*288 (K-padded el)
    float* Wp1    = ws + 1439232;     // 288*128
    float* Wp2    = ws + 1476096;     // 288*128
    // end: 1512960 floats (~6.05 MB)

    float* score  = out;
    float* eu_out = out + (size_t)NU * NI;                       // NU*272
    float* el_out = out + (size_t)NU * NI + (size_t)NU * 272;    // NI*272

    const int ne_u = NU * 32;
    const int ne_v = NI * 32;
    const int* src_u = ug_u_edges;            const int* dst_u = ug_u_edges + ne_u;
    const int* src_v = ug_v_edges;            const int* dst_v = ug_v_edges + ne_v;

    // ---- attention + s2d ----
    att_kernel<<<(NU * NI) / 256, 256, 0, stream>>>(u_v_idx, ui_table, att_W1, att_b1,
                                                    att_W2, att_b2, w_att);
    softmax_s2d_kernel<<<NU, 256, 0, stream>>>(w_att, u_v_idx, ui_table,
                                               user_idx, user_table, s2d);

    // ---- degrees ----
    deg_init_kernel<<<(NU + 255) / 256, 256, 0, stream>>>(dinv_u, NU);
    deg_add_kernel <<<(ne_u + 255) / 256, 256, 0, stream>>>(dst_u, dinv_u, ne_u);
    deg_fin_kernel <<<(NU + 255) / 256, 256, 0, stream>>>(dinv_u, NU);
    deg_init_kernel<<<(NI + 255) / 256, 256, 0, stream>>>(dinv_v, NI);
    deg_add_kernel <<<(ne_v + 255) / 256, 256, 0, stream>>>(dst_v, dinv_v, ne_v);
    deg_fin_kernel <<<(NI + 255) / 256, 256, 0, stream>>>(dinv_v, NI);

    // ---- pe gather (into pre_v region, consumed before pre_v is produced) ----
    float* pe = pre_v;
    gather_pe_kernel<<<(NI * ND) / 256, 256, 0, stream>>>(item_idx, item_table, pe);

    // ---- user GCN stack ----
    gemm_wmma_f16<<<dim3(NU / 16, 64 / 16), 32, 0, stream>>>(s2d, 64, ugcn_W1, 64, htmp, 64, 64);
    agg_init_kernel<<<(NU * 64) / 256, 256, 0, stream>>>(htmp, dinv_u, eu1, NU, 64);
    agg_edge_kernel<<<(ne_u * 64) / 256, 256, 0, stream>>>(htmp, src_u, dst_u, dinv_u, eu1, ne_u, 64);
    agg_fin_kernel <<<(NU * 64) / 256, 256, 0, stream>>>(eu1, ugcn_b1, NU, 64);

    gemm_wmma_f16<<<dim3(NU / 16, 32 / 16), 32, 0, stream>>>(eu1, 64, ugcn_W2, 32, htmp, 32, 64);
    agg_init_kernel<<<(NU * 32) / 256, 256, 0, stream>>>(htmp, dinv_u, eu2, NU, 32);
    agg_edge_kernel<<<(ne_u * 32) / 256, 256, 0, stream>>>(htmp, src_u, dst_u, dinv_u, eu2, ne_u, 32);
    agg_fin_kernel <<<(NU * 32) / 256, 256, 0, stream>>>(eu2, ugcn_b2, NU, 32);

    gemm_wmma_f16<<<dim3(NU / 16, 16 / 16), 32, 0, stream>>>(eu2, 32, ugcn_W3, 16, htmp, 16, 32);
    agg_init_kernel<<<(NU * 16) / 256, 256, 0, stream>>>(htmp, dinv_u, eu3, NU, 16);
    agg_edge_kernel<<<(ne_u * 16) / 256, 256, 0, stream>>>(htmp, src_u, dst_u, dinv_u, eu3, ne_u, 16);
    agg_fin_kernel <<<(NU * 16) / 256, 256, 0, stream>>>(eu3, ugcn_b3, NU, 16);

    // ---- item GCN stack ----
    gemm_wmma_f16<<<dim3(NI / 16, 64 / 16), 32, 0, stream>>>(pe, 64, igcn_W1, 64, htmp, 64, 64);
    agg_init_kernel<<<(NI * 64) / 256, 256, 0, stream>>>(htmp, dinv_v, el1, NI, 64);
    agg_edge_kernel<<<(ne_v * 64) / 256, 256, 0, stream>>>(htmp, src_v, dst_v, dinv_v, el1, ne_v, 64);
    agg_fin_kernel <<<(NI * 64) / 256, 256, 0, stream>>>(el1, igcn_b1, NI, 64);

    gemm_wmma_f16<<<dim3(NI / 16, 32 / 16), 32, 0, stream>>>(el1, 64, igcn_W2, 32, htmp, 32, 64);
    agg_init_kernel<<<(NI * 32) / 256, 256, 0, stream>>>(htmp, dinv_v, el2, NI, 32);
    agg_edge_kernel<<<(ne_v * 32) / 256, 256, 0, stream>>>(htmp, src_v, dst_v, dinv_v, el2, ne_v, 32);
    agg_fin_kernel <<<(NI * 32) / 256, 256, 0, stream>>>(el2, igcn_b2, NI, 32);

    gemm_wmma_f16<<<dim3(NI / 16, 16 / 16), 32, 0, stream>>>(el2, 32, igcn_W3, 16, htmp, 16, 32);
    agg_init_kernel<<<(NI * 16) / 256, 256, 0, stream>>>(htmp, dinv_v, el3, NI, 16);
    agg_edge_kernel<<<(ne_v * 16) / 256, 256, 0, stream>>>(htmp, src_v, dst_v, dinv_v, el3, ne_v, 16);
    agg_fin_kernel <<<(NI * 16) / 256, 256, 0, stream>>>(el3, igcn_b3, NI, 16);

    // ---- concat (padded ws copy + dense d_out copy) ----
    concat_pad_kernel<<<(NU * 288 + 255) / 256, 256, 0, stream>>>(eu1, s2d, eu2, eu3, euP, eu_out, NU);
    concat_pad_kernel<<<(NI * 288 + 255) / 256, 256, 0, stream>>>(el1, pe, el2, el3, elP, el_out, NI);

    // ---- pad lin1_W halves ----
    pad_w_kernel<<<(288 * 128 + 255) / 256, 256, 0, stream>>>(lin1_W, Wp1, Wp2);

    // ---- lin1 projections (K = 288, branch-free) ----
    gemm_wmma_f16<<<dim3(NU / 16, 128 / 16), 32, 0, stream>>>(euP, 288, Wp1, 128, pre_u, 128, 288);
    gemm_wmma_f16<<<dim3(NI / 16, 128 / 16), 32, 0, stream>>>(elP, 288, Wp2, 128, pre_v, 128, 288);

    // ---- lin2 weights to f16 ----
    cvt_f16_kernel<<<(128 * 64 + 255) / 256, 256, 0, stream>>>(lin2_W, lin2h, 128 * 64);

    // ---- big pair MLP: 512 users x 64 item-tiles, 4 waves/block ----
    pair_mlp_kernel<<<dim3(NU, NI / 64), 128, 0, stream>>>(pre_u, pre_v, lin1_b, lin2h,
                                                           lin2_b, out_W, out_b, score);
}
